// MLP_21354577396166
// MI455X (gfx1250) — matrix operand order repaired and checked by reading, not compiled, not run on hardware
//
#include <hip/hip_runtime.h>
#include <stdint.h>

#define T_TOK 8192
#define DDIM  1024
#define FDIM  4096
#define NEXP  4
#define BM    64          // tokens per block

// LDS strides (bf16 elements), padded for conflict-free 16B fragment reads
#define XSTR  1032
#define WKSTR 136         // gate/up staged tiles [64 f][128 k]
#define HSTR  72          // H tile [64 m][64 f]
#define WDSTR 72          // down staged tile [256 d][64 f]

#define WGBUF (64 * WKSTR * 2)      // 17408 B per gate/up buffer
#define WDBUF (256 * WDSTR * 2)     // 36864 B per down buffer

// LDS byte offsets (dynamic-LDS arena starts at offset 0); double-buffered
#define XS_OFF   0
#define WG_OFF   (XS_OFF + BM * XSTR * 2)            // 132096
#define WU_OFF   (WG_OFF + 2 * WGBUF)                // 166912
#define HS_OFF   (WU_OFF + 2 * WGBUF)                // 201728
#define WD_OFF   (HS_OFF + BM * HSTR * 2)            // 210944
#define CMB_OFF  (WD_OFF + 2 * WDBUF)                // 284672
#define LDS_BYTES (CMB_OFF + BM * NEXP * 4)          // 285696 (< 320 KB WGP)

// workspace byte offsets
#define CMB_WS 0
#define XB_WS  (T_TOK * NEXP * 4)                                  // 131072
#define WG_WS  (XB_WS + (size_t)T_TOK * DDIM * 2)
#define WU_WS  (WG_WS + (size_t)NEXP * DDIM * FDIM * 2)
#define WD_WS  (WU_WS + (size_t)NEXP * DDIM * FDIM * 2)

typedef __attribute__((ext_vector_type(16))) __bf16   bf16x16;
typedef __attribute__((ext_vector_type(4)))  __bf16   bf16x4;
typedef __attribute__((ext_vector_type(8)))  float    floatx8;
typedef __attribute__((ext_vector_type(4)))  uint32_t uintx4;

struct Frag32 { uintx4 lo; uintx4 hi; };

// 16-byte global -> LDS async copy (ASYNCcnt).  VDST = per-lane LDS byte
// offset (hardware adds the wave's LDS base); VADDR = 64-bit global address.
__device__ __forceinline__ void async_copy16(const void* g, int lds_off) {
  asm volatile("global_load_async_to_lds_b128 %0, %1, off"
               :: "v"(lds_off), "v"((unsigned long long)g)
               : "memory");
}
// Drain all own async copies
__device__ __forceinline__ void wait_async0() {
  asm volatile("s_wait_asynccnt 0x0" ::: "memory");
}
// Allow the 8 most recent copies (next stage) to stay in flight
__device__ __forceinline__ void wait_async8() {
  asm volatile("s_wait_asynccnt 0x8" ::: "memory");
}

// A fragment (16x32 bf16): lanes 0-15 rows, K chunks {0..7,16..23};
// lanes 16-31 rows, K chunks {8..15,24..31}
__device__ __forceinline__ bf16x16 ldsA(const __bf16* base, int stride, int row0, int k0, int lane) {
  const int half = lane >> 4;
  const __bf16* p = base + (row0 + (lane & 15)) * stride + k0 + half * 8;
  Frag32 fr;
  fr.lo = *(const uintx4*)(p);
  fr.hi = *(const uintx4*)(p + 16);
  return __builtin_bit_cast(bf16x16, fr);
}

// B fragment (32x16 bf16) from [n][k] tile: lanes 0-15 K {0..7,16..23}, lanes 16-31 K {8..15,24..31}
__device__ __forceinline__ bf16x16 ldsB(const __bf16* base, int stride, int n0, int k0, int lane) {
  const int half = lane >> 4;
  const __bf16* p = base + (n0 + (lane & 15)) * stride + k0 + half * 8;
  Frag32 fr;
  fr.lo = *(const uintx4*)(p);
  fr.hi = *(const uintx4*)(p + 16);
  return __builtin_bit_cast(bf16x16, fr);
}

__device__ __forceinline__ floatx8 wmma_bf16(bf16x16 a, bf16x16 b, floatx8 c) {
  return __builtin_amdgcn_wmma_f32_16x16x32_bf16(false, a, false, b, (short)0, c, false, false);
}

// ---------------------------------------------------------------------------
// Pre-pass: x fp32 -> bf16
// ---------------------------------------------------------------------------
__global__ __launch_bounds__(256) void cvt_x_kernel(
    const float* __restrict__ x, __bf16* __restrict__ xb)
{
  size_t i = ((size_t)blockIdx.x * 256 + threadIdx.x) * 4;
  const float4 v = *(const float4*)(x + i);
  bf16x4 o = {(__bf16)v.x, (__bf16)v.y, (__bf16)v.z, (__bf16)v.w};
  *(bf16x4*)(xb + i) = o;
}

// ---------------------------------------------------------------------------
// Pre-pass: per-expert transpose + fp32->bf16.  src [E][R][C] -> dst [E][C][R]
// ---------------------------------------------------------------------------
__global__ __launch_bounds__(256) void transpose_cvt_kernel(
    const float* __restrict__ src, __bf16* __restrict__ dst, int R, int C)
{
  __shared__ float tile[32][33];
  const float* s = src + (size_t)blockIdx.z * R * C;
  __bf16*      d = dst + (size_t)blockIdx.z * R * C;
  int c0 = blockIdx.x * 32, r0 = blockIdx.y * 32;
  int tx = threadIdx.x, ty = threadIdx.y;
  #pragma unroll
  for (int j = 0; j < 32; j += 8)
    tile[ty + j][tx] = s[(size_t)(r0 + ty + j) * C + c0 + tx];
  __syncthreads();
  #pragma unroll
  for (int j = 0; j < 32; j += 8)
    d[(size_t)(c0 + ty + j) * R + r0 + tx] = (__bf16)tile[tx][ty + j];
}

// ---------------------------------------------------------------------------
// Router: wave-per-token; combine[T,E] (top-2 renormalized), logits to d_out.
// ---------------------------------------------------------------------------
__global__ __launch_bounds__(256) void router_kernel(
    const float* __restrict__ x, const float* __restrict__ Wr,
    float* __restrict__ logits, float* __restrict__ combine)
{
  const int wave = threadIdx.x >> 5, lane = threadIdx.x & 31;
  const int t = blockIdx.x * 8 + wave;
  const float* xr = x + (size_t)t * DDIM;
  float s0 = 0.f, s1 = 0.f, s2 = 0.f, s3 = 0.f;
  for (int d = lane; d < DDIM; d += 32) {
    float xv = xr[d];
    s0 += xv * Wr[0 * DDIM + d];
    s1 += xv * Wr[1 * DDIM + d];
    s2 += xv * Wr[2 * DDIM + d];
    s3 += xv * Wr[3 * DDIM + d];
  }
  #pragma unroll
  for (int off = 16; off > 0; off >>= 1) {
    s0 += __shfl_xor(s0, off, 32);
    s1 += __shfl_xor(s1, off, 32);
    s2 += __shfl_xor(s2, off, 32);
    s3 += __shfl_xor(s3, off, 32);
  }
  float s[4] = {s0, s1, s2, s3};
  float mx = fmaxf(fmaxf(s0, s1), fmaxf(s2, s3));
  float p[4]; float ps = 0.f;
  #pragma unroll
  for (int e = 0; e < 4; ++e) { p[e] = __expf(s[e] - mx); ps += p[e]; }
  #pragma unroll
  for (int e = 0; e < 4; ++e) p[e] /= ps;
  int i1 = 0;
  #pragma unroll
  for (int e = 1; e < 4; ++e) if (p[e] > p[i1]) i1 = e;
  int i2 = (i1 == 0) ? 1 : 0;
  #pragma unroll
  for (int e = 0; e < 4; ++e) if (e != i1 && p[e] > p[i2]) i2 = e;
  float den = p[i1] + p[i2];
  float c[4] = {0.f, 0.f, 0.f, 0.f};
  c[i1] = p[i1] / den;
  c[i2] = p[i2] / den;
  if (lane < 4) {
    logits [(size_t)t * NEXP + lane] = s[lane];
    combine[(size_t)t * NEXP + lane] = c[lane];
  }
}

// ---------------------------------------------------------------------------
// Fused MoE MLP on pre-transposed bf16 weights.
// Block = 64 tokens x full 1024-dim output in registers (32 acc tiles/wave).
// Weight staging = double-buffered 16B async copies; s_wait_asynccnt 0x8
// leaves the next stage's 8 copies in flight behind the WMMA burst.
// ---------------------------------------------------------------------------
__global__ __launch_bounds__(256, 1) void moe_fused_kernel(
    const __bf16* __restrict__ xb,  const __bf16* __restrict__ wgT,
    const __bf16* __restrict__ wuT, const __bf16* __restrict__ wdT,
    const float* __restrict__ combine, float* __restrict__ out)
{
  extern __shared__ char smem[];
  __bf16* XS = (__bf16*)(smem + XS_OFF);   // [64][XSTR]
  __bf16* HS = (__bf16*)(smem + HS_OFF);   // [64][HSTR]
  float*  CMB = (float*)(smem + CMB_OFF);  // [64][4]

  const int tid  = threadIdx.x;
  const int wave = tid >> 5, lane = tid & 31;
  const int mi   = wave & 3;        // M tile 0..3 (rows mi*16..)
  const int grp  = wave >> 2;       // N parity group 0..1
  const int t0   = blockIdx.x * BM;
  const int rsel = (lane >> 4) << 3;

  // stage X tile (bf16, already converted) via async copies; reused 256x
  for (int ii = 0; ii < 32; ++ii) {
    int i  = (ii << 8) + tid;        // 0..8191
    int r  = i >> 7;                 // row 0..63
    int c8 = (i & 127) << 3;         // bf16 col, multiple of 8
    async_copy16(xb + (size_t)(t0 + r) * DDIM + c8,
                 XS_OFF + (r * XSTR + c8) * 2);
  }
  CMB[tid] = combine[(size_t)t0 * NEXP + tid];   // 64 tokens x 4 experts = 256
  wait_async0();
  __syncthreads();

  floatx8 acc[32];
  const floatx8 fzero = {0.f,0.f,0.f,0.f,0.f,0.f,0.f,0.f};
  #pragma unroll
  for (int i = 0; i < 32; ++i) acc[i] = fzero;

  for (int e = 0; e < NEXP; ++e) {
    const __bf16* wg = wgT + (size_t)e * FDIM * DDIM;   // [F][D]
    const __bf16* wu = wuT + (size_t)e * FDIM * DDIM;   // [F][D]
    const __bf16* wd = wdT + (size_t)e * DDIM * FDIM;   // [D][F]
    for (int fc = 0; fc < FDIM; fc += 64) {
      // 8 asyncs/wave: one 128-K gate+up stage into buffer `buf`
      auto stage_gu = [&](int ks0, int buf) {
        const int bo = buf * WGBUF;
        for (int ii = 0; ii < 4; ++ii) {
          int i  = (ii << 8) + tid;     // 0..1023
          int f  = i >> 4;              // 0..63
          int k8 = (i & 15) << 3;       // 0..120
          async_copy16(wg + (size_t)(fc + f) * DDIM + ks0 + k8,
                       WG_OFF + bo + (f * WKSTR + k8) * 2);
          async_copy16(wu + (size_t)(fc + f) * DDIM + ks0 + k8,
                       WU_OFF + bo + (f * WKSTR + k8) * 2);
        }
      };
      // 8 asyncs/wave: one 256-col down-proj stage into buffer `buf`
      auto stage_wd = [&](int nh, int buf) {
        const int bo = buf * WDBUF;
        for (int ii = 0; ii < 8; ++ii) {
          int i  = (ii << 8) + tid;     // 0..2047
          int dl = i >> 3;              // 0..255
          int f8 = (i & 7) << 3;        // 0..56
          async_copy16(wd + (size_t)(nh * 256 + dl) * FDIM + fc + f8,
                       WD_OFF + bo + (dl * WDSTR + f8) * 2);
        }
      };

      // ---- GEMM1: wave computes tiles (mi, f-tiles grp*2 / grp*2+1), g & u
      floatx8 g0 = fzero, g1 = fzero, u0 = fzero, u1 = fzero;
      stage_gu(0, 0);
      for (int s = 0; s < 8; ++s) {           // 8 K-stages of 128
        __syncthreads();                       // readers of buf[(s+1)&1] done
        if (s < 7) { stage_gu((s + 1) << 7, (s + 1) & 1); wait_async8(); }
        else       { wait_async0(); }
        __syncthreads();                       // stage s published
        const __bf16* WGb = (const __bf16*)(smem + WG_OFF + (s & 1) * WGBUF);
        const __bf16* WUb = (const __bf16*)(smem + WU_OFF + (s & 1) * WGBUF);
        const int ks0 = s << 7;
        #pragma unroll
        for (int kk = 0; kk < 4; ++kk) {
          bf16x16 a   = ldsA(XS, XSTR, mi * 16, ks0 + kk * 32, lane);
          bf16x16 bg0 = ldsB(WGb, WKSTR, (grp * 2 + 0) * 16, kk * 32, lane);
          bf16x16 bg1 = ldsB(WGb, WKSTR, (grp * 2 + 1) * 16, kk * 32, lane);
          bf16x16 bu0 = ldsB(WUb, WKSTR, (grp * 2 + 0) * 16, kk * 32, lane);
          bf16x16 bu1 = ldsB(WUb, WKSTR, (grp * 2 + 1) * 16, kk * 32, lane);
          g0 = wmma_bf16(a, bg0, g0);
          g1 = wmma_bf16(a, bg1, g1);
          u0 = wmma_bf16(a, bu0, u0);
          u1 = wmma_bf16(a, bu1, u1);
        }
      }
      // ---- epilogue: H = silu(g)*u * combine[t,e] -> LDS bf16
      #pragma unroll
      for (int t = 0; t < 2; ++t) {
        floatx8 g = t ? g1 : g0;
        floatx8 u = t ? u1 : u0;
        int col = (grp * 2 + t) * 16 + (lane & 15);
        #pragma unroll
        for (int j = 0; j < 8; ++j) {
          int m = mi * 16 + rsel + j;
          float cw = CMB[m * NEXP + e];
          float gg = g[j], uu = u[j];
          HS[m * HSTR + col] = (__bf16)((gg / (1.0f + __expf(-gg))) * uu * cw);
        }
      }
      __syncthreads();                         // publish HS
      // ---- GEMM2: acc[64x1024] += H[64x64] @ WdT^T, N in 256-col quarters
      bf16x16 aH0 = ldsA(HS, HSTR, mi * 16, 0,  lane);
      bf16x16 aH1 = ldsA(HS, HSTR, mi * 16, 32, lane);
      stage_wd(0, 0);
      for (int nh = 0; nh < 4; ++nh) {
        __syncthreads();                       // readers of buf[(nh+1)&1] done
        if (nh < 3) { stage_wd(nh + 1, (nh + 1) & 1); wait_async8(); }
        else        { wait_async0(); }
        __syncthreads();                       // stage nh published
        const __bf16* WDb = (const __bf16*)(smem + WD_OFF + (nh & 1) * WDBUF);
        #pragma unroll
        for (int ks = 0; ks < 2; ++ks) {
          bf16x16 aH = ks ? aH1 : aH0;
          #pragma unroll
          for (int j = 0; j < 8; ++j) {
            bf16x16 b = ldsB(WDb, WDSTR, (2 * j + grp) * 16, ks * 32, lane);
            acc[nh * 8 + j] = wmma_bf16(aH, b, acc[nh * 8 + j]);
          }
        }
      }
    }
  }

  // ---- write output (fp32)
  #pragma unroll
  for (int a2 = 0; a2 < 32; ++a2) {
    int nh = a2 >> 3, j = a2 & 7;
    int ncol = (nh * 16 + 2 * j + grp) * 16 + (lane & 15);
    #pragma unroll
    for (int jj = 0; jj < 8; ++jj)
      out[(size_t)(t0 + mi * 16 + rsel + jj) * DDIM + ncol] = acc[a2][jj];
  }
}

extern "C" void kernel_launch(void* const* d_in, const int* in_sizes, int n_in,
                              void* d_out, int out_size, void* d_ws, size_t ws_size,
                              hipStream_t stream) {
  const float* x  = (const float*)d_in[0];
  const float* Wr = (const float*)d_in[1];
  const float* Wg = (const float*)d_in[2];
  const float* Wu = (const float*)d_in[3];
  const float* Wd = (const float*)d_in[4];
  float* out    = (float*)d_out;
  float* logits = out + (size_t)T_TOK * DDIM;       // tuple output #2

  char* ws = (char*)d_ws;
  float*  combine = (float*)(ws + CMB_WS);
  __bf16* xb      = (__bf16*)(ws + XB_WS);
  __bf16* wgT     = (__bf16*)(ws + WG_WS);
  __bf16* wuT     = (__bf16*)(ws + WU_WS);
  __bf16* wdT     = (__bf16*)(ws + WD_WS);

  cvt_x_kernel<<<(T_TOK * DDIM) / 1024, 256, 0, stream>>>(x, xb);
  transpose_cvt_kernel<<<dim3(FDIM / 32, DDIM / 32, NEXP), dim3(32, 8), 0, stream>>>(Wg, wgT, DDIM, FDIM);
  transpose_cvt_kernel<<<dim3(FDIM / 32, DDIM / 32, NEXP), dim3(32, 8), 0, stream>>>(Wu, wuT, DDIM, FDIM);
  transpose_cvt_kernel<<<dim3(DDIM / 32, FDIM / 32, NEXP), dim3(32, 8), 0, stream>>>(Wd, wdT, FDIM, DDIM);
  router_kernel<<<T_TOK / 8, 256, 0, stream>>>(x, Wr, logits, combine);
  moe_fused_kernel<<<T_TOK / BM, 256, LDS_BYTES, stream>>>(xb, wgT, wuT, wdT, combine, out);
}
